// my_test_model_38036230374107
// MI455X (gfx1250) — compile-verified, matrix operand order checked by reading, not run on hardware
//
#include <hip/hip_runtime.h>
#include <hip/hip_bf16.h>

// ---------------- types ----------------
typedef __attribute__((ext_vector_type(16))) _Float16 v16h;
typedef __attribute__((ext_vector_type(8)))  _Float16 v8h;
typedef __attribute__((ext_vector_type(8)))  float    v8f;

#define BATCH   128
#define GRID_N  28
#define HID     64
#define NDIR    4
#define FIVEH   320   // 5*H
#define TWOH    128   // 2*H
#define FC_HD   512
#define OUTC    10

// CDNA5 async global->LDS path (ASYNCcnt), guarded so either toolchain compiles.
#if defined(__HIP_DEVICE_COMPILE__) && __has_builtin(__builtin_amdgcn_global_load_async_to_lds_b128) && __has_builtin(__builtin_amdgcn_s_wait_asynccnt)
#define HAVE_ASYNC_LDS 1
#else
#define HAVE_ASYNC_LDS 0
#endif

#if HAVE_ASYNC_LDS
// builtin prototype (from hipcc diagnostic): (v4i addrspace(1)*, v4i addrspace(3)*, imm i32, imm i32)
typedef int v4i_t __attribute__((vector_size(16)));
typedef __attribute__((address_space(1))) v4i_t* gptr_v4i;
typedef __attribute__((address_space(3))) v4i_t* lptr_v4i;
#endif

// ---------------- helpers ----------------
// Load a 16x32 fp16 WMMA A-style fragment from a row-major [rows][rowstride] fp16 array.
// Per ISA 7.12.2 (16-bit A 16x32): lanes 0-15 hold M=0..15 K={kb..kb+7, kb+16..kb+23},
// lanes 16-31 hold M=0..15 K={kb+8..kb+15, kb+24..kb+31}.
__device__ __forceinline__ v16h load_frag16(const _Float16* base, int row, int rowstride,
                                            int kbase, int lane) {
  int mrow = row + (lane & 15);
  int koff = kbase + ((lane >> 4) << 3);
  const _Float16* p = base + (size_t)mrow * rowstride + koff;
  v8h lo = *(const v8h*)(p);
  v8h hi = *(const v8h*)(p + 16);
  v16h r;
#pragma unroll
  for (int q = 0; q < 8; ++q) { r[q] = lo[q]; r[q + 8] = hi[q]; }
  return r;
}

__device__ __forceinline__ v16h zero_frag16() {
  v16h r;
#pragma unroll
  for (int q = 0; q < 16; ++q) r[q] = (_Float16)0.0f;
  return r;
}

__device__ __forceinline__ float sigm(float v) { return 1.0f / (1.0f + __expf(-v)); }
// tanh(v) = 2*sigmoid(2v) - 1  (single v_exp_f32, saturates cleanly at +-1)
__device__ __forceinline__ float tanh_fast(float v) {
  return 2.0f / (1.0f + __expf(-2.0f * v)) - 1.0f;
}

// ---------------- weight conversion ----------------
// U: [4][128][320] f32 -> Uh: [4][320][128] fp16 (N-major for B fragments)
__global__ void convert_U_kernel(const float* __restrict__ U, _Float16* __restrict__ Uh) {
  int i = blockIdx.x * 256 + threadIdx.x;
  if (i >= NDIR * TWOH * FIVEH) return;
  int dd  = i / (TWOH * FIVEH);
  int rem = i % (TWOH * FIVEH);
  int k   = rem / FIVEH;
  int n   = rem % FIVEH;
  Uh[((size_t)dd * FIVEH + n) * TWOH + k] = (_Float16)U[i];
}

// W1: [256][512] f32 -> W1T: [512][256] fp16
__global__ void convert_W1_kernel(const float* __restrict__ W1, _Float16* __restrict__ W1T) {
  int i = blockIdx.x * 256 + threadIdx.x;
  if (i >= (NDIR * HID) * FC_HD) return;
  int k = i / FC_HD;
  int n = i % FC_HD;
  W1T[(size_t)n * (NDIR * HID) + k] = (_Float16)W1[i];
}

// ---------------- MD-LSTM anti-diagonal step ----------------
// grid: (diag_len, 4 dirs), block: 256 threads = 8 waves.
// hdiag: fp16 [4][2][28][128][64], cdiag: f32 same shape. Double-buffered on diag parity.
// B (the per-direction U weights, 80 KB fp16) is staged once per WG into LDS so all
// 8 waves share it; A fragments come straight from global (h_left/h_up never straddle
// a fragment, so no A staging is needed).
__global__ __launch_bounds__(256) void mdlstm_diag_kernel(
    const float* __restrict__ x,        // [128][28][28]
    const float* __restrict__ Wx,       // [4][320]
    const float* __restrict__ bias,     // [4][320]
    const _Float16* __restrict__ Uh,    // [4][320][128]
    _Float16* __restrict__ hdiag,
    float* __restrict__ cdiag,
    float* __restrict__ hfinal,         // [4][128][64]
    int d, int r0)
{
  __shared__ _Float16 Blds[FIVEH * TWOH];   // 80 KB: U for this direction, N-major
  __shared__ float    xpix[BATCH];

  const int dd = blockIdx.y;
  const int r  = r0 + blockIdx.x;
  const int c  = d - r;
  const int tid = threadIdx.x;

  // ---- stage pixel column for this cell (apply direction flips) ----
  const int rr = (dd & 2) ? (GRID_N - 1 - r) : r;
  const int cc = (dd & 1) ? (GRID_N - 1 - c) : c;
  for (int m = tid; m < BATCH; m += 256)
    xpix[m] = x[(size_t)m * (GRID_N * GRID_N) + rr * GRID_N + cc];

  // ---- stage B = U[dd] (320x128 fp16) into LDS, 16B chunks ----
  const _Float16* Ub = Uh + (size_t)dd * FIVEH * TWOH;
#pragma unroll 4
  for (int idx = tid; idx < (FIVEH * TWOH) / 8; idx += 256) {
#if HAVE_ASYNC_LDS
    __builtin_amdgcn_global_load_async_to_lds_b128(
        (gptr_v4i)(Ub + (size_t)idx * 8),
        (lptr_v4i)(Blds + (size_t)idx * 8),
        0, 0);
#else
    *(v8h*)(Blds + (size_t)idx * 8) = *(const v8h*)(Ub + (size_t)idx * 8);
#endif
  }
#if HAVE_ASYNC_LDS
  __builtin_amdgcn_s_wait_asynccnt(0);
#endif
  __syncthreads();

  // ---- per-wave A fragments straight from global h-state ----
  const int wv    = tid >> 5;
  const int lane  = tid & 31;
  const int mbase = wv * 16;

  const int pprev = (d - 1) & 1;
  const int rm1   = (r > 0) ? (r - 1) : 0;
  const _Float16* hl = hdiag + ((((size_t)dd * 2 + pprev) * GRID_N + r  ) * BATCH) * HID;
  const _Float16* hu = hdiag + ((((size_t)dd * 2 + pprev) * GRID_N + rm1) * BATCH) * HID;

  v16h afrag[4];  // K = [h_left(0..63) | h_up(64..127)], fragments never straddle
  if (c > 0) {
    afrag[0] = load_frag16(hl, mbase, HID, 0,  lane);
    afrag[1] = load_frag16(hl, mbase, HID, 32, lane);
  } else {
    afrag[0] = zero_frag16();
    afrag[1] = zero_frag16();
  }
  if (r > 0) {
    afrag[2] = load_frag16(hu, mbase, HID, 0,  lane);
    afrag[3] = load_frag16(hu, mbase, HID, 32, lane);
  } else {
    afrag[2] = zero_frag16();
    afrag[3] = zero_frag16();
  }

  const int pcur = d & 1;
  const float* cl = cdiag + ((((size_t)dd * 2 + pprev) * GRID_N + r  ) * BATCH) * HID;
  const float* cu = cdiag + ((((size_t)dd * 2 + pprev) * GRID_N + rm1) * BATCH) * HID;
  float*    cw = cdiag + ((((size_t)dd * 2 + pcur) * GRID_N + r) * BATCH) * HID;
  _Float16* hw = hdiag + ((((size_t)dd * 2 + pcur) * GRID_N + r) * BATCH) * HID;

#pragma unroll
  for (int j = 0; j < 4; ++j) {            // hidden chunk of 16 columns
    v8f accs[5];
#pragma unroll
    for (int g = 0; g < 5; ++g) {          // gates i, f1, f2, o, g
      v8f acc = {};
      int n0 = g * HID + j * 16;
#pragma unroll
      for (int kk = 0; kk < 4; ++kk) {
        v16h bfrag = load_frag16(Blds, n0, TWOH, kk * 32, lane);
        acc = __builtin_amdgcn_wmma_f32_16x16x32_f16(false, afrag[kk], false, bfrag,
                                                     (short)0, acc, false, false);
      }
      accs[g] = acc;
    }

    // epilogue on the 16x16 f32 C layout: lane l, slot v -> (m = 16w + v + 8*(l>=16), n = l&15)
    const int ncol = j * 16 + (lane & 15);
    const int mofs = (lane >> 4) << 3;
    float wxv[5], bv[5];
#pragma unroll
    for (int g = 0; g < 5; ++g) {
      wxv[g] = Wx[dd * FIVEH + g * HID + ncol];
      bv[g]  = bias[dd * FIVEH + g * HID + ncol];
    }
#pragma unroll
    for (int v = 0; v < 8; ++v) {
      int   m  = mbase + mofs + v;
      float xp = xpix[m];
      float zi  = accs[0][v] + xp * wxv[0] + bv[0];
      float zf1 = accs[1][v] + xp * wxv[1] + bv[1];
      float zf2 = accs[2][v] + xp * wxv[2] + bv[2];
      float zo  = accs[3][v] + xp * wxv[3] + bv[3];
      float zg  = accs[4][v] + xp * wxv[4] + bv[4];
      float clft = (c > 0) ? cl[(size_t)m * HID + ncol] : 0.0f;
      float cup  = (r > 0) ? cu[(size_t)m * HID + ncol] : 0.0f;
      float cn = sigm(zi) * tanh_fast(zg) + sigm(zf1) * clft + sigm(zf2) * cup;
      float hn = sigm(zo) * tanh_fast(cn);
      cw[(size_t)m * HID + ncol] = cn;
      hw[(size_t)m * HID + ncol] = (_Float16)hn;
      if (r == GRID_N - 1 && c == GRID_N - 1)
        hfinal[((size_t)dd * BATCH + m) * HID + ncol] = hn;
    }
  }
}

// ---------------- FC1 (relu(h@W1+b1)) via WMMA: [128x256]x[256x512] ----------------
__global__ __launch_bounds__(256) void fc1_kernel(
    const float* __restrict__ hfinal,   // [4][128][64]
    const _Float16* __restrict__ W1T,   // [512][256]
    const float* __restrict__ b1,       // [512]
    _Float16* __restrict__ z1)          // [128][512]
{
  __shared__ _Float16 Alds[BATCH * (NDIR * HID)];  // 64 KB, h concat per batch
  const int tid = threadIdx.x;
  for (int i = tid; i < BATCH * NDIR * HID; i += 256) {
    int m   = i >> 8;
    int col = i & 255;
    int dd  = col >> 6;
    int hh  = col & 63;
    Alds[i] = (_Float16)hfinal[((size_t)dd * BATCH + m) * HID + hh];
  }
  __syncthreads();

  const int wv = tid >> 5, lane = tid & 31;
  const int mbase = wv * 16;
  v16h afrag[8];
#pragma unroll
  for (int kk = 0; kk < 8; ++kk)
    afrag[kk] = load_frag16(Alds, mbase, NDIR * HID, kk * 32, lane);

  for (int n = 0; n < FC_HD / 16; ++n) {
    v8f acc = {};
#pragma unroll
    for (int kk = 0; kk < 8; ++kk) {
      v16h bfrag = load_frag16(W1T, n * 16, NDIR * HID, kk * 32, lane);
      acc = __builtin_amdgcn_wmma_f32_16x16x32_f16(false, afrag[kk], false, bfrag,
                                                   (short)0, acc, false, false);
    }
    const int ncol = n * 16 + (lane & 15);
    const int mofs = (lane >> 4) << 3;
    float bb = b1[ncol];
#pragma unroll
    for (int v = 0; v < 8; ++v) {
      int m = mbase + mofs + v;
      float zz = acc[v] + bb;
      z1[(size_t)m * FC_HD + ncol] = (_Float16)fmaxf(zz, 0.0f);
    }
  }
}

// ---------------- FC2 + softmax ----------------
__global__ __launch_bounds__(32) void fc2_softmax_kernel(
    const _Float16* __restrict__ z1,   // [128][512]
    const float* __restrict__ W2,      // [512][10]
    const float* __restrict__ b2,      // [10]
    float* __restrict__ out)           // [128][10]
{
  __shared__ float logits[OUTC];
  const int m = blockIdx.x;
  const int t = threadIdx.x;
  if (t < OUTC) {
    float s = b2[t];
    for (int k = 0; k < FC_HD; ++k)
      s += (float)z1[(size_t)m * FC_HD + k] * W2[(size_t)k * OUTC + t];
    logits[t] = s;
  }
  __syncthreads();
  if (t == 0) {
    float mx = logits[0];
    for (int i = 1; i < OUTC; ++i) mx = fmaxf(mx, logits[i]);
    float e[OUTC], sum = 0.0f;
    for (int i = 0; i < OUTC; ++i) { e[i] = __expf(logits[i] - mx); sum += e[i]; }
    float inv = 1.0f / sum;
    for (int i = 0; i < OUTC; ++i) out[(size_t)m * OUTC + i] = e[i] * inv;
  }
}

// ---------------- launch ----------------
extern "C" void kernel_launch(void* const* d_in, const int* in_sizes, int n_in,
                              void* d_out, int out_size, void* d_ws, size_t ws_size,
                              hipStream_t stream) {
  const float* x   = (const float*)d_in[0];
  const float* Wx  = (const float*)d_in[1];
  const float* U   = (const float*)d_in[2];
  const float* b   = (const float*)d_in[3];
  const float* W1  = (const float*)d_in[4];
  const float* b1  = (const float*)d_in[5];
  const float* W2  = (const float*)d_in[6];
  const float* b2  = (const float*)d_in[7];
  float* out = (float*)d_out;

  // workspace carve-up (all offsets 16B aligned)
  char* ws = (char*)d_ws;
  _Float16* Uh    = (_Float16*)ws; ws += (size_t)NDIR * FIVEH * TWOH * sizeof(_Float16);          // 320 KB
  _Float16* W1T   = (_Float16*)ws; ws += (size_t)FC_HD * NDIR * HID * sizeof(_Float16);           // 256 KB
  _Float16* hdiag = (_Float16*)ws; ws += (size_t)NDIR * 2 * GRID_N * BATCH * HID * sizeof(_Float16);
  float*    cdiag = (float*)ws;    ws += (size_t)NDIR * 2 * GRID_N * BATCH * HID * sizeof(float);
  float*    hfin  = (float*)ws;    ws += (size_t)NDIR * BATCH * HID * sizeof(float);
  _Float16* z1    = (_Float16*)ws; ws += (size_t)BATCH * FC_HD * sizeof(_Float16);

  {
    int n = NDIR * TWOH * FIVEH;
    convert_U_kernel<<<(n + 255) / 256, 256, 0, stream>>>(U, Uh);
  }
  {
    int n = NDIR * HID * FC_HD;
    convert_W1_kernel<<<(n + 255) / 256, 256, 0, stream>>>(W1, W1T);
  }

  // anti-diagonal wavefront: 55 dependent steps, each fully parallel across cells & dirs
  for (int d = 0; d < 2 * GRID_N - 1; ++d) {
    int r0  = d - (GRID_N - 1); if (r0 < 0) r0 = 0;
    int r1  = d < (GRID_N - 1) ? d : (GRID_N - 1);
    int len = r1 - r0 + 1;
    mdlstm_diag_kernel<<<dim3(len, NDIR), 256, 0, stream>>>(
        x, Wx, b, Uh, hdiag, cdiag, hfin, d, r0);
  }

  fc1_kernel<<<1, 256, 0, stream>>>(hfin, W1T, b1, z1);
  fc2_softmax_kernel<<<BATCH, 32, 0, stream>>>(z1, W2, b2, out);
}
